// MTP_Mask_Attention_88416196756292
// MI455X (gfx1250) — matrix operand order repaired and checked by reading, not compiled, not run on hardware
//
#include <hip/hip_runtime.h>
#include <hip/hip_bf16.h>

typedef __attribute__((ext_vector_type(16))) __bf16 v16bf;
typedef __attribute__((ext_vector_type(8)))  float  v8f;
typedef __attribute__((ext_vector_type(4)))  unsigned int v4u;

#define TSEQ   128
#define DMODEL 512
#define NHEADS 8
#define DH     64
#define QKV_ELEMS (512u * 8u * 128u * 64u)   // 33,554,432 bf16 elems per tensor
#define WT_LD  40                             // 32 K + 8 pad (ushorts) -> 80B rows, 16B aligned
#define VT_LD  136                            // 128 T + 8 pad

// ---------------- TDM availability probes ----------------
#ifndef USE_TDM
#if defined(__has_builtin)
#if __has_builtin(__builtin_amdgcn_tensor_load_to_lds) && __has_builtin(__builtin_amdgcn_s_wait_tensorcnt)
#define USE_TDM 1
#endif
#endif
#endif
#ifndef USE_TDM
#define USE_TDM 0
#endif

__device__ __forceinline__ unsigned short f32_to_bf16(float f) {
  unsigned int u = __float_as_uint(f);
  u += 0x7FFFu + ((u >> 16) & 1u);          // round-to-nearest-even
  return (unsigned short)(u >> 16);
}

__device__ __forceinline__ unsigned int pack_bf16x2(float lo, float hi) {
  return (unsigned int)f32_to_bf16(lo) | ((unsigned int)f32_to_bf16(hi) << 16);
}

// Build a bf16 fragment from two 16-byte LDS chunks (forces ds_load_b128 pairs)
__device__ __forceinline__ v16bf frag_ld2(const unsigned short* p0, const unsigned short* p1) {
  union { v4u u[2]; v16bf b; } c;
  c.u[0] = *(const v4u*)p0;
  c.u[1] = *(const v4u*)p1;
  return c.b;
}

// A fragment: 16(M) x 32(K), row-major tile with leading dim ld (ushorts).
// lanes 0-15 -> M=lane, K {0..7,16..23}; lanes 16-31 -> M=lane-16, K {8..15,24..31}
__device__ __forceinline__ v16bf frag_a(const unsigned short* base, int ld, int row0, int k0, int lane) {
  const unsigned short* p = base + (size_t)(row0 + (lane & 15)) * ld + k0 + ((lane >> 4) << 3);
  return frag_ld2(p, p + 16);
}

// B fragment: 32(K) x 16(N) sourced from an [N][K] (transposed/packed) tile, leading dim ld.
// lanes 0-15 -> K {0..7,16..23}, lanes 16-31 -> K {8..15,24..31}; N = lane&15
__device__ __forceinline__ v16bf frag_bp(const unsigned short* base, int ld, int n0, int k0, int lane) {
  const unsigned short* p = base + (size_t)(n0 + (lane & 15)) * ld + k0 + ((lane >> 4) << 3);
  return frag_ld2(p, p + 16);
}

// ---------------- TDM helpers ----------------
#if USE_TDM
typedef unsigned int u32x4 __attribute__((ext_vector_type(4)));
typedef int i32x4 __attribute__((ext_vector_type(4)));
typedef int i32x8 __attribute__((ext_vector_type(8)));

__device__ __forceinline__ unsigned lds_raw_offset(const void* p) {
  return (unsigned)(unsigned long long)
      (__attribute__((address_space(3))) const char*)(const char*)p;
}

// 2D tensor load: rows x cols bf16 elements, contiguous row-major, to LDS.
__device__ __forceinline__ void tdm_load_2d(const void* gptr, const void* lptr,
                                            unsigned rows, unsigned cols) {
  const unsigned long long ga = (unsigned long long)gptr;
  u32x4 g0;
  g0[0] = 1u;                                                  // count=1, user descriptor
  g0[1] = lds_raw_offset(lptr);                                // lds_addr
  g0[2] = (unsigned)(ga & 0xFFFFFFFFu);                        // global_addr[31:0]
  g0[3] = (unsigned)((ga >> 32) & 0x01FFFFFFu) | (2u << 30);   // global_addr[56:32] | type=2
  i32x8 g1;
  g1[0] = 1 << 16;                                             // data_size=1 (2 bytes/elem)
  g1[1] = (int)((cols & 0xFFFFu) << 16);                       // tensor_dim0[15:0]
  g1[2] = (int)((cols >> 16) | ((rows & 0xFFFFu) << 16));      // td0[31:16] | tensor_dim1[15:0]
  g1[3] = (int)((rows >> 16) | (cols << 16));                  // td1[31:16] | tile_dim0
  g1[4] = (int)rows;                                           // tile_dim1 (tile_dim2 = 0)
  g1[5] = (int)cols;                                           // tensor_dim0_stride[31:0]
  g1[6] = 0;
  g1[7] = 0;
  i32x4 z4 = {0, 0, 0, 0};
#if __has_include(<hip/amd_detail/amd_gfx1250_TDM.h>)
  i32x8 z8 = {0, 0, 0, 0, 0, 0, 0, 0};
  __builtin_amdgcn_tensor_load_to_lds(g0, g1, z4, z4, z8, 0);
#else
  __builtin_amdgcn_tensor_load_to_lds(g0, g1, z4, z4, 0);
#endif
}
#endif

// ---------------- Kernel 1: fp32 -> bf16 weight conversion ----------------
__global__ __launch_bounds__(256)
void cvt_weights(const float* __restrict__ wq, const float* __restrict__ wk,
                 const float* __restrict__ wv, const float* __restrict__ wo,
                 unsigned short* __restrict__ wqb, unsigned short* __restrict__ wkb,
                 unsigned short* __restrict__ wvb, unsigned short* __restrict__ wob) {
  const int i = blockIdx.x * 256 + threadIdx.x;   // 0 .. 262143
  wqb[i] = f32_to_bf16(wq[i]);
  wkb[i] = f32_to_bf16(wk[i]);
  wvb[i] = f32_to_bf16(wv[i]);
  wob[i] = f32_to_bf16(wo[i]);
}

// ---------------- Kernel 2: fused Q/K/V projection GEMM ----------------
// grid = (M/128, D/64); block = 256 (8 waves). One x tile feeds 3 weight matrices.
__global__ __launch_bounds__(256)
void qkv_gemm(const float* __restrict__ x,
              const unsigned short* __restrict__ wqb, const unsigned short* __restrict__ wkb,
              const unsigned short* __restrict__ wvb,
              const float* __restrict__ bq, const float* __restrict__ bk, const float* __restrict__ bv,
              unsigned short* __restrict__ qo, unsigned short* __restrict__ ko,
              unsigned short* __restrict__ vo) {
  __shared__ __align__(16) unsigned short Xs[128 * 32];        // [m][k]
  __shared__ __align__(16) unsigned short Wt[3][64 * WT_LD];   // transposed [n][k]
  const int tid  = threadIdx.x;
  const int lane = tid & 31;
  const int wave = tid >> 5;
  const int m0 = blockIdx.x * 128;
  const int n0 = blockIdx.y * 64;
  const unsigned short* wptr[3] = {wqb, wkb, wvb};

  v8f acc[3][4] = {};

  for (int k0 = 0; k0 < DMODEL; k0 += 32) {
    __syncthreads();
    // X tile: float4 loads, packed bf16 stores
    for (int i = tid; i < 1024; i += 256) {            // 1024 float4 = 4096 elems
      const int r  = i >> 3;                           // 8 float4 per 32-wide row
      const int c4 = (i & 7) << 2;
      const float4 v = *(const float4*)(x + (size_t)(m0 + r) * DMODEL + k0 + c4);
      unsigned int* dst = (unsigned int*)(Xs + r * 32 + c4);
      dst[0] = pack_bf16x2(v.x, v.y);
      dst[1] = pack_bf16x2(v.z, v.w);
    }
    // W tiles: uint loads over N, transposed stores into [n][k]
#pragma unroll
    for (int m = 0; m < 3; ++m)
      for (int i = tid; i < 1024; i += 256) {          // 1024 uint = 2048 elems
        const int kk = i >> 5;
        const int n2 = (i & 31) << 1;
        const unsigned int v =
            *(const unsigned int*)(wptr[m] + (size_t)(k0 + kk) * DMODEL + n0 + n2);
        Wt[m][n2 * WT_LD + kk]       = (unsigned short)v;
        Wt[m][(n2 + 1) * WT_LD + kk] = (unsigned short)(v >> 16);
      }
    if (k0 + 32 < DMODEL)
      __builtin_prefetch(x + (size_t)(m0 + (tid >> 1)) * DMODEL + k0 + 32, 0, 1);
    __syncthreads();

    const v16bf a = frag_a(Xs, 32, wave * 16, 0, lane);
#pragma unroll
    for (int m = 0; m < 3; ++m) {
      v16bf b[4];
#pragma unroll
      for (int j = 0; j < 4; ++j) b[j] = frag_bp(Wt[m], WT_LD, j * 16, 0, lane);
#pragma unroll
      for (int j = 0; j < 4; ++j)
        acc[m][j] = __builtin_amdgcn_wmma_f32_16x16x32_bf16(
            false, a, false, b[j], (short)0, acc[m][j], false, false);
    }
  }

  // epilogue: bias add, bf16 cast, scatter into [BN, H, T, Dh] layout
  const int half = (lane >> 4) & 1;
  const int lc   = lane & 15;
  const float* bias[3] = {bq, bk, bv};
  unsigned short* outp[3] = {qo, ko, vo};
#pragma unroll
  for (int m = 0; m < 3; ++m)
#pragma unroll
    for (int j = 0; j < 4; ++j) {
      const int gn = n0 + j * 16 + lc;
      const float bb = bias[m][gn];
      const int h = gn >> 6, dh = gn & 63;
#pragma unroll
      for (int r = 0; r < 8; ++r) {
        const int gm = m0 + wave * 16 + r + (half << 3);
        const int bn = gm >> 7, t = gm & 127;
        outp[m][((size_t)(bn * NHEADS + h) * TSEQ + t) * DH + dh] =
            f32_to_bf16(acc[m][j][r] + bb);
      }
    }
}

// ---------------- Kernel 3: per-(bn,h) masked attention ----------------
// grid = BN*H = 4096; block = 256 (8 waves, wave w owns query rows 16w..16w+15)
__global__ __launch_bounds__(256)
void attention(const unsigned short* __restrict__ qws, const unsigned short* __restrict__ kws,
               const unsigned short* __restrict__ vws, const float* __restrict__ mask,
               unsigned short* __restrict__ ows) {
  __shared__ __align__(16) unsigned char smem[116224];
  unsigned short* Qs = (unsigned short*)(smem);             // 16 KB  [t][dh]
  unsigned short* Ks = (unsigned short*)(smem + 16384);     // 16 KB  [t][dh]
  unsigned short* Vt = (unsigned short*)(smem + 32768);     // 17 KB  [dh][t] (transposed, ld 136)
  float*          S  = (float*)(smem + 50176);              // 64 KB  [q][k]
  float*          mr = (float*)(smem + 115712);             // 512 B
  unsigned short* P  = (unsigned short*)(smem);             // 32 KB, overlays Qs+Ks

  const int tid  = threadIdx.x;
  const int lane = tid & 31;
  const int wave = tid >> 5;
  const int bn = blockIdx.x >> 3;
  const int h  = blockIdx.x & 7;
  const size_t base = ((size_t)(bn * NHEADS + h)) * TSEQ * DH;

#if USE_TDM
  if (wave == 0) {
    tdm_load_2d(qws + base, Qs, TSEQ, DH);
    tdm_load_2d(kws + base, Ks, TSEQ, DH);
    __builtin_amdgcn_s_wait_tensorcnt(0);
  }
#else
  {
    const v4u* qsrc = (const v4u*)(qws + base);
    const v4u* ksrc = (const v4u*)(kws + base);
    v4u* qdst = (v4u*)Qs;
    v4u* kdst = (v4u*)Ks;
    for (int i = tid; i < 1024; i += 256) {   // 1024 x 16B = 16 KB each
      qdst[i] = qsrc[i];
      kdst[i] = ksrc[i];
    }
  }
#endif
  // V: load [t][dh], store transposed [dh][t]
  {
    const unsigned int* vsrc = (const unsigned int*)(vws + base);
    for (int i = tid; i < 4096; i += 256) {   // 4096 uints = 8192 elems
      const int t  = i >> 5;
      const int d2 = (i & 31) << 1;
      const unsigned int v = vsrc[i];
      Vt[d2 * VT_LD + t]       = (unsigned short)v;
      Vt[(d2 + 1) * VT_LD + t] = (unsigned short)(v >> 16);
    }
  }
  if (tid < TSEQ) mr[tid] = mask[bn * TSEQ + tid];
  __syncthreads();

  // S = (Q K^T) * 1/sqrt(Dh);  B of K^T = [n][k] view of K -> frag_bp on Ks
  const float scale = 0.125f;
  const int q0   = wave * 16;
  const int half = (lane >> 4) & 1;
  const int lc   = lane & 15;
  {
    const v16bf qa0 = frag_a(Qs, DH, q0, 0,  lane);   // reused across all 8 tiles
    const v16bf qa1 = frag_a(Qs, DH, q0, 32, lane);
#pragma unroll
    for (int j = 0; j < 8; ++j) {
      const v16bf b0 = frag_bp(Ks, DH, j * 16, 0,  lane);
      const v16bf b1 = frag_bp(Ks, DH, j * 16, 32, lane);
      v8f acc = {};
      acc = __builtin_amdgcn_wmma_f32_16x16x32_bf16(
          false, qa0, false, b0, (short)0, acc, false, false);
      acc = __builtin_amdgcn_wmma_f32_16x16x32_bf16(
          false, qa1, false, b1, (short)0, acc, false, false);
#pragma unroll
      for (int r = 0; r < 8; ++r)
        S[(q0 + r + (half << 3)) * TSEQ + j * 16 + lc] = acc[r] * scale;
    }
  }
  __syncthreads();

  // masked softmax, one query row per thread (fp32, float4-vectorized)
  if (tid < TSEQ) {
    const int q = tid;
    unsigned int* prow = (unsigned int*)(P + q * TSEQ);
    if (mr[q] == 0.0f) {
      for (int k = 0; k < TSEQ / 2; ++k) prow[k] = 0u;
    } else {
      float m = -3.0e38f;
      for (int k = 0; k < TSEQ; k += 4) {
        const float4 s = *(const float4*)(S + q * TSEQ + k);
        const float4 mm = *(const float4*)(mr + k);
        if (mm.x != 0.0f && s.x > m) m = s.x;
        if (mm.y != 0.0f && s.y > m) m = s.y;
        if (mm.z != 0.0f && s.z > m) m = s.z;
        if (mm.w != 0.0f && s.w > m) m = s.w;
      }
      float sum = 0.0f;
      for (int k = 0; k < TSEQ; k += 4) {
        float4 s = *(const float4*)(S + q * TSEQ + k);
        const float4 mm = *(const float4*)(mr + k);
        s.x = (mm.x != 0.0f) ? __expf(s.x - m) : 0.0f;
        s.y = (mm.y != 0.0f) ? __expf(s.y - m) : 0.0f;
        s.z = (mm.z != 0.0f) ? __expf(s.z - m) : 0.0f;
        s.w = (mm.w != 0.0f) ? __expf(s.w - m) : 0.0f;
        *(float4*)(S + q * TSEQ + k) = s;
        sum += s.x + s.y + s.z + s.w;
      }
      const float inv = sum > 0.0f ? 1.0f / sum : 0.0f;
      for (int k = 0; k < TSEQ; k += 4) {
        const float4 s = *(const float4*)(S + q * TSEQ + k);
        prow[(k >> 1) + 0] = pack_bf16x2(s.x * inv, s.y * inv);
        prow[(k >> 1) + 1] = pack_bf16x2(s.z * inv, s.w * inv);
      }
    }
  }
  __syncthreads();

  // O = P V  -> ows[(bn*T + t)*D + h*Dh + n]; B frags from transposed Vt
  {
    v16bf pa[4];
#pragma unroll
    for (int kk = 0; kk < 4; ++kk) pa[kk] = frag_a(P, TSEQ, q0, kk * 32, lane);
#pragma unroll
    for (int j = 0; j < 4; ++j) {
      v16bf vb[4];
#pragma unroll
      for (int kk = 0; kk < 4; ++kk) vb[kk] = frag_bp(Vt, VT_LD, j * 16, kk * 32, lane);
      v8f acc = {};
#pragma unroll
      for (int kk = 0; kk < 4; ++kk)
        acc = __builtin_amdgcn_wmma_f32_16x16x32_bf16(
            false, pa[kk], false, vb[kk], (short)0, acc, false, false);
#pragma unroll
      for (int r = 0; r < 8; ++r) {
        const int t = q0 + r + (half << 3);
        ows[((size_t)bn * TSEQ + t) * DMODEL + h * DH + j * 16 + lc] = f32_to_bf16(acc[r]);
      }
    }
  }
}

// ---------------- Kernel 4: output projection (fp32 result) ----------------
__global__ __launch_bounds__(256)
void out_gemm(const unsigned short* __restrict__ o, const unsigned short* __restrict__ wob,
              const float* __restrict__ bias, float* __restrict__ out) {
  __shared__ __align__(16) unsigned short Xs[128 * 32];
  __shared__ __align__(16) unsigned short Wt[64 * WT_LD];
  const int tid  = threadIdx.x;
  const int lane = tid & 31;
  const int wave = tid >> 5;
  const int m0 = blockIdx.x * 128;
  const int n0 = blockIdx.y * 64;

  v8f acc[4] = {};
  for (int k0 = 0; k0 < DMODEL; k0 += 32) {
    __syncthreads();
    for (int i = tid; i < 512; i += 256) {            // 512 x 16B = 8 KB
      const int r  = i >> 2;
      const int c8 = (i & 3) << 3;
      *(v4u*)(Xs + r * 32 + c8) =
          *(const v4u*)(o + (size_t)(m0 + r) * DMODEL + k0 + c8);
    }
    for (int i = tid; i < 1024; i += 256) {
      const int kk = i >> 5;
      const int n2 = (i & 31) << 1;
      const unsigned int v =
          *(const unsigned int*)(wob + (size_t)(k0 + kk) * DMODEL + n0 + n2);
      Wt[n2 * WT_LD + kk]       = (unsigned short)v;
      Wt[(n2 + 1) * WT_LD + kk] = (unsigned short)(v >> 16);
    }
    if (k0 + 32 < DMODEL)
      __builtin_prefetch(o + (size_t)(m0 + (tid >> 1)) * DMODEL + k0 + 32, 0, 1);
    __syncthreads();

    const v16bf a = frag_a(Xs, 32, wave * 16, 0, lane);
    v16bf b[4];
#pragma unroll
    for (int j = 0; j < 4; ++j) b[j] = frag_bp(Wt, WT_LD, j * 16, 0, lane);
#pragma unroll
    for (int j = 0; j < 4; ++j)
      acc[j] = __builtin_amdgcn_wmma_f32_16x16x32_bf16(
          false, a, false, b[j], (short)0, acc[j], false, false);
  }
  const int half = (lane >> 4) & 1;
  const int lc   = lane & 15;
#pragma unroll
  for (int j = 0; j < 4; ++j) {
    const int gn = n0 + j * 16 + lc;
    const float bb = bias[gn];
#pragma unroll
    for (int r = 0; r < 8; ++r) {
      const int gm = m0 + wave * 16 + r + (half << 3);
      out[(size_t)gm * DMODEL + gn] = acc[j][r] + bb;
    }
  }
}

extern "C" void kernel_launch(void* const* d_in, const int* in_sizes, int n_in,
                              void* d_out, int out_size, void* d_ws, size_t ws_size,
                              hipStream_t stream) {
  const float* x    = (const float*)d_in[0];
  const float* mask = (const float*)d_in[1];
  const float* Wq   = (const float*)d_in[2];
  const float* bq   = (const float*)d_in[3];
  const float* Wk   = (const float*)d_in[4];
  const float* bk   = (const float*)d_in[5];
  const float* Wv   = (const float*)d_in[6];
  const float* bv   = (const float*)d_in[7];
  const float* Wo   = (const float*)d_in[8];
  const float* bo   = (const float*)d_in[9];

  unsigned short* wqb = (unsigned short*)d_ws;
  unsigned short* wkb = wqb + 262144;
  unsigned short* wvb = wkb + 262144;
  unsigned short* wob = wvb + 262144;
  unsigned short* qws = wob + 262144;
  unsigned short* kws = qws + QKV_ELEMS;
  unsigned short* vws = kws + QKV_ELEMS;
  unsigned short* ows = vws + QKV_ELEMS;

  cvt_weights<<<1024, 256, 0, stream>>>(Wq, Wk, Wv, Wo, wqb, wkb, wvb, wob);

  dim3 gproj(512, 8, 1);   // M/128 x N/64
  qkv_gemm<<<gproj, 256, 0, stream>>>(x, wqb, wkb, wvb, bq, bk, bv, qws, kws, vws);

  attention<<<4096, 256, 0, stream>>>(qws, kws, vws, mask, ows);

  out_gemm<<<gproj, 256, 0, stream>>>(ows, wob, bo, (float*)d_out);
}
